// BiLSTM_CRF_61942018343714
// MI455X (gfx1250) — matrix operand order repaired and checked by reading, not compiled, not run on hardware
//
#include <hip/hip_runtime.h>
#include <hip/hip_bf16.h>

typedef __attribute__((ext_vector_type(16))) __bf16 v16bf;
typedef __attribute__((ext_vector_type(8)))  float  v8f;

#define EMB    256
#define HID    512
#define BATCH  128
#define SEQT   256
#define KTAG   12
#define TSTART 10
#define TSTOP  11
#define G4     2048   // 4*HID
#define KCAT   768    // EMB+HID
#define KTILES 24     // KCAT/32
#define NTILES 128    // G4/16
#define NWG    32     // workgroups per direction
#define NTH    256

// ---------------------------------------------------------------- fast activations (branch-free)
__device__ __forceinline__ float sigmoid_f(float x) {
    return __builtin_amdgcn_rcpf(1.0f + __expf(-x));
}
__device__ __forceinline__ float tanh_f(float x) {
    // 1 - 2/(e^{2x}+1); exp->inf => +1, exp->0 => -1 (saturates correctly, no branches)
    float e = __expf(2.0f * x);
    return 1.0f - 2.0f * __builtin_amdgcn_rcpf(e + 1.0f);
}

// ---------------------------------------------------------------- barriers
__device__ __forceinline__ void grid_barrier(int* ctr, int target) {
    __syncthreads();
    if (threadIdx.x == 0) {
        __threadfence();
        atomicAdd(ctr, 1);
        while (__hip_atomic_load(ctr, __ATOMIC_RELAXED, __HIP_MEMORY_SCOPE_AGENT) < target) {
            __builtin_amdgcn_s_sleep(2);
        }
        __threadfence();
    }
    __syncthreads();
}

__global__ void zero_ctr(int* ctr) {
    if (threadIdx.x < 128) ctr[threadIdx.x] = 0;
}

// ---------------------------------------------------------------- prep
__global__ void embed_kernel(const int* __restrict__ x,
                             const float* __restrict__ emb,
                             __bf16* __restrict__ xs) {
    int flat = blockIdx.x * 256 + threadIdx.x;
    if (flat >= SEQT * BATCH * EMB) return;
    int e = flat & 255;
    int b = (flat >> 8) & 127;
    int t = flat >> 15;
    int row = x[b * SEQT + t];
    xs[flat] = (__bf16)emb[(size_t)row * EMB + e];
}

// WMMA B-fragment layout: element i of lane L holds B[K = kt*32 + (i>=8?16:0) + (L/16)*8 + (i&7)][N = nt*16 + (L%16)]
// where B[k][g] = Wcat[g][k] = (k<EMB ? W_ih[g][k] : W_hh[g][k-EMB]).
__global__ void prep_wtfrag(const float* __restrict__ W_ih_f, const float* __restrict__ W_hh_f,
                            const float* __restrict__ W_ih_b, const float* __restrict__ W_hh_b,
                            __bf16* __restrict__ out) {
    const int perdir = KTILES * NTILES * 512;  // 1,572,864
    int flat = blockIdx.x * 256 + threadIdx.x;
    if (flat >= 2 * perdir) return;
    int dir = flat / perdir;
    int r   = flat - dir * perdir;
    int kt  = r / (NTILES * 512);
    int r2  = r - kt * (NTILES * 512);
    int nt  = r2 >> 9;
    int r3  = r2 & 511;
    int lane = r3 >> 4;
    int i    = r3 & 15;
    int k = kt * 32 + ((i & 8) << 1) + (lane >> 4) * 8 + (i & 7);
    int n = nt * 16 + (lane & 15);
    const float* W_ih = dir ? W_ih_b : W_ih_f;
    const float* W_hh = dir ? W_hh_b : W_hh_f;
    float v = (k < EMB) ? W_ih[(size_t)n * EMB + k] : W_hh[(size_t)n * HID + (k - EMB)];
    out[flat] = (__bf16)v;
}

__global__ void prep_bias(const float* __restrict__ bif, const float* __restrict__ bhf,
                          const float* __restrict__ bib, const float* __restrict__ bhb,
                          float* __restrict__ bias) {
    int g = blockIdx.x * 256 + threadIdx.x;
    if (g < G4)            bias[g] = bif[g] + bhf[g];
    else if (g < 2 * G4)   bias[g] = bib[g - G4] + bhb[g - G4];
}

// ---------------------------------------------------------------- LSTM (persistent, both dirs)
// Each wave owns (mt, jt): M-rows mt*16..+16, hidden cols jt*16..+16, and computes the
// i/f/g/o blocks for those columns (N-tiles jt, jt+32, jt+64, jt+96). The LSTM cell is
// fused in-register; c and carried h live in VGPRs for the whole sequence. h is published
// as bf16 into a double-buffered array (read s&1, write (s+1)&1) -> 1 grid barrier/step.
__global__ void __launch_bounds__(NTH) lstm_kernel(
        const __bf16* __restrict__ xs, const __bf16* __restrict__ wtfrag,
        const float* __restrict__ bias_all, __bf16* __restrict__ hbf_all,
        __bf16* __restrict__ hs_all, const int* __restrict__ lengths,
        const float* __restrict__ h0, const float* __restrict__ c0,
        int* __restrict__ ctr_all) {
    const int dir = (blockIdx.x >= NWG) ? 1 : 0;
    const int wg  = blockIdx.x - dir * NWG;
    const int tid = threadIdx.x;

    const __bf16* wt = wtfrag + (size_t)dir * (KTILES * NTILES * 512);
    const float*  bs = bias_all + (size_t)dir * G4;
    __bf16* hb0 = hbf_all + (size_t)dir * (2 * BATCH * HID);
    __bf16* hb1 = hb0 + BATCH * HID;
    __bf16* hs  = hs_all + (size_t)dir * ((size_t)SEQT * BATCH * HID);
    int* ctr = ctr_all + dir * 64;

    // publish initial h (bf16) into buffer 0
    for (int idx = wg * NTH + tid; idx < BATCH * HID; idx += NWG * NTH)
        hb0[idx] = (__bf16)h0[(size_t)dir * BATCH * HID + idx];
    int gen = 1;
    grid_barrier(ctr, gen * NWG); gen++;

    const int lane  = tid & 31;
    const int wv    = tid >> 5;
    const int gwave = wg * 8 + wv;       // 0..255
    const int mt    = gwave >> 5;        // 0..7  (batch tile)
    const int jt    = gwave & 31;        // 0..31 (hidden-column tile)
    const int mrow  = mt * 16 + (lane & 15);
    const int kb    = (lane >> 4) * 8;   // A-layout half select
    const int j     = jt * 16 + (lane & 15);
    const int mbase = mt * 16 + (lane >> 4) * 8;

    // per-lane loop-invariant state
    const float bi  = bs[j];
    const float bff = bs[HID + j];
    const float bg  = bs[2 * HID + j];
    const float bo  = bs[3 * HID + j];
    int   len[8];
    float cst[8], hst[8];
#pragma unroll
    for (int v = 0; v < 8; ++v) {
        len[v] = lengths[mbase + v];
        cst[v] = c0[(size_t)dir * BATCH * HID + (size_t)(mbase + v) * HID + j];
        hst[v] = h0[(size_t)dir * BATCH * HID + (size_t)(mbase + v) * HID + j];
    }

    for (int s = 0; s < SEQT; ++s) {
        const int t = dir ? (SEQT - 1 - s) : s;
        const __bf16* hread  = (s & 1) ? hb1 : hb0;
        __bf16*       hwrite = (s & 1) ? hb0 : hb1;
        const __bf16* xrow = xs    + (size_t)(t * BATCH + mrow) * EMB;
        const __bf16* hrow = hread + (size_t)mrow * HID;

        v8f acc0 = {0.f,0.f,0.f,0.f,0.f,0.f,0.f,0.f};
        v8f acc1 = acc0, acc2 = acc0, acc3 = acc0;

        for (int kt = 0; kt < KTILES; ++kt) {
            // ---- A fragment (16-bit A 16x32 layout): elem i -> K = kofs + (i>=8?16:0) + kb + (i&7)
            const __bf16* src;
            int kofs;
            if (kt < 8) { src = xrow; kofs = kt * 32; }
            else        { src = hrow; kofs = kt * 32 - EMB; }
            v16bf afrag;
#pragma unroll
            for (int i = 0; i < 16; ++i) {
                int kk = kofs + ((i & 8) << 1) + kb + (i & 7);
                afrag[i] = src[kk];
            }
            // ---- B fragments for the i/f/g/o column blocks (pre-swizzled, 32B vector loads)
            const __bf16* wbase = wt + ((size_t)(kt * NTILES + jt) * 32 + lane) * 16;
            v16bf bf0 = *(const v16bf*)(wbase);                      // i block (nt = jt)
            v16bf bf1 = *(const v16bf*)(wbase + 32 * 512);           // f block (nt = jt+32)
            v16bf bf2 = *(const v16bf*)(wbase + 64 * 512);           // g block (nt = jt+64)
            v16bf bf3 = *(const v16bf*)(wbase + 96 * 512);           // o block (nt = jt+96)
            acc0 = __builtin_amdgcn_wmma_f32_16x16x32_bf16(false, afrag, false, bf0, (short)0, acc0, false, false);
            acc1 = __builtin_amdgcn_wmma_f32_16x16x32_bf16(false, afrag, false, bf1, (short)0, acc1, false, false);
            acc2 = __builtin_amdgcn_wmma_f32_16x16x32_bf16(false, afrag, false, bf2, (short)0, acc2, false, false);
            acc3 = __builtin_amdgcn_wmma_f32_16x16x32_bf16(false, afrag, false, bf3, (short)0, acc3, false, false);
        }

        // ---- fused LSTM cell, fully in registers (element v -> batch row mbase+v, col j)
#pragma unroll
        for (int v = 0; v < 8; ++v) {
            float iv = acc0[v] + bi;
            float fv = acc1[v] + bff;
            float gv = acc2[v] + bg;
            float ov = acc3[v] + bo;
            float si = sigmoid_f(iv);
            float sf = sigmoid_f(fv);
            float so = sigmoid_f(ov);
            float tg = tanh_f(gv);
            float cn = sf * cst[v] + si * tg;
            float hn = so * tanh_f(cn);
            bool msk = (t < len[v]);
            cst[v] = msk ? cn : cst[v];
            hst[v] = msk ? hn : hst[v];
            __bf16 hbv = (__bf16)hst[v];
            hwrite[(size_t)(mbase + v) * HID + j] = hbv;
            hs[(size_t)(t * BATCH + mbase + v) * HID + j] = hbv;
        }
        __threadfence();
        grid_barrier(ctr, gen * NWG); gen++;
    }
}

// ---------------------------------------------------------------- emissions (K=12, VALU)
__global__ void emissions_kernel(const __bf16* __restrict__ hs_f, const __bf16* __restrict__ hs_b,
                                 const float* __restrict__ fc_W, const float* __restrict__ fc_b,
                                 float* __restrict__ E) {
    int r = blockIdx.x * 256 + threadIdx.x;  // t*BATCH + b
    if (r >= SEQT * BATCH) return;
    float acc[KTAG];
#pragma unroll
    for (int k = 0; k < KTAG; ++k) acc[k] = fc_b[k];
    const __bf16* hf = hs_f + (size_t)r * HID;
    const __bf16* hb = hs_b + (size_t)r * HID;
    for (int j = 0; j < HID; ++j) {
        float v = (float)hf[j];
#pragma unroll
        for (int k = 0; k < KTAG; ++k) acc[k] += v * fc_W[k * (2 * HID) + j];
    }
    for (int j = 0; j < HID; ++j) {
        float v = (float)hb[j];
#pragma unroll
        for (int k = 0; k < KTAG; ++k) acc[k] += v * fc_W[k * (2 * HID) + HID + j];
    }
#pragma unroll
    for (int k = 0; k < KTAG; ++k) E[(size_t)r * KTAG + k] = acc[k];
}

// ---------------------------------------------------------------- CRF: logZ - gold
__global__ void crf_kernel(const float* __restrict__ E, const int* __restrict__ lengths,
                           const int* __restrict__ tags, const float* __restrict__ trans,
                           float* __restrict__ out) {
    __shared__ float tr[KTAG * KTAG];
    __shared__ float alpha[KTAG];
    int b = blockIdx.x;
    int lane = threadIdx.x;
    for (int i = lane; i < KTAG * KTAG; i += 32) tr[i] = trans[i];
    if (lane < KTAG) alpha[lane] = (lane == TSTART) ? 0.0f : -10000.0f;
    __syncthreads();
    int len = lengths[b];
    for (int t = 0; t < SEQT; ++t) {
        float nv = 0.0f;
        if (lane < KTAG) {
            float mx = -1e30f;
            float sc[KTAG];
#pragma unroll
            for (int j = 0; j < KTAG; ++j) { sc[j] = alpha[j] + tr[lane * KTAG + j]; mx = fmaxf(mx, sc[j]); }
            float sum = 0.0f;
#pragma unroll
            for (int j = 0; j < KTAG; ++j) sum += __expf(sc[j] - mx);
            nv = mx + __logf(sum) + E[(size_t)(t * BATCH + b) * KTAG + lane];
        }
        __syncthreads();
        if (lane < KTAG && t < len) alpha[lane] = nv;
        __syncthreads();
    }
    if (lane == 0) {
        float mx = -1e30f, v[KTAG];
#pragma unroll
        for (int k = 0; k < KTAG; ++k) { v[k] = alpha[k] + tr[TSTOP * KTAG + k]; mx = fmaxf(mx, v[k]); }
        float sum = 0.0f;
#pragma unroll
        for (int k = 0; k < KTAG; ++k) sum += __expf(v[k] - mx);
        float logZ = mx + __logf(sum);
        float gold = 0.0f;
        int prev = TSTART;
        for (int t = 0; t < len; ++t) {
            int nt_ = tags[b * SEQT + t];
            gold += tr[nt_ * KTAG + prev] + E[(size_t)(t * BATCH + b) * KTAG + nt_];
            prev = nt_;
        }
        gold += tr[TSTOP * KTAG + prev];
        out[b] = logZ - gold;
    }
}

// ---------------------------------------------------------------- launch
extern "C" void kernel_launch(void* const* d_in, const int* in_sizes, int n_in,
                              void* d_out, int out_size, void* d_ws, size_t ws_size,
                              hipStream_t stream) {
    const int*   x         = (const int*)d_in[0];
    const int*   lengths   = (const int*)d_in[1];
    const int*   tags      = (const int*)d_in[2];
    const float* embedding = (const float*)d_in[3];
    const float* W_ih_f    = (const float*)d_in[4];
    const float* W_hh_f    = (const float*)d_in[5];
    const float* b_ih_f    = (const float*)d_in[6];
    const float* b_hh_f    = (const float*)d_in[7];
    const float* W_ih_b    = (const float*)d_in[8];
    const float* W_hh_b    = (const float*)d_in[9];
    const float* b_ih_b    = (const float*)d_in[10];
    const float* b_hh_b    = (const float*)d_in[11];
    const float* fc_W      = (const float*)d_in[12];
    const float* fc_b      = (const float*)d_in[13];
    const float* trans     = (const float*)d_in[14];
    const float* h0        = (const float*)d_in[15];
    const float* c0        = (const float*)d_in[16];
    float* out = (float*)d_out;

    char* ws = (char*)d_ws;
    size_t off = 0;
    auto alloc = [&](size_t bytes) -> char* {
        char* p = ws + off;
        off = (off + bytes + 255) & ~(size_t)255;
        return p;
    };
    __bf16* xs     = (__bf16*)alloc((size_t)SEQT * BATCH * EMB * 2);
    __bf16* wtfrag = (__bf16*)alloc((size_t)2 * KTILES * NTILES * 512 * 2);
    float*  bias   = (float*)alloc((size_t)2 * G4 * 4);
    __bf16* hbf    = (__bf16*)alloc((size_t)2 * 2 * BATCH * HID * 2);  // double-buffered per dir
    __bf16* hs     = (__bf16*)alloc((size_t)2 * SEQT * BATCH * HID * 2);
    float*  E      = (float*)alloc((size_t)SEQT * BATCH * KTAG * 4);
    int*    ctr    = (int*)alloc(512);

    zero_ctr<<<1, 128, 0, stream>>>(ctr);
    embed_kernel<<<(SEQT * BATCH * EMB + 255) / 256, 256, 0, stream>>>(x, embedding, xs);
    prep_wtfrag<<<(2 * KTILES * NTILES * 512 + 255) / 256, 256, 0, stream>>>(
        W_ih_f, W_hh_f, W_ih_b, W_hh_b, wtfrag);
    prep_bias<<<(2 * G4 + 255) / 256, 256, 0, stream>>>(b_ih_f, b_hh_f, b_ih_b, b_hh_b, bias);
    lstm_kernel<<<2 * NWG, NTH, 0, stream>>>(xs, wtfrag, bias, hbf, hs, lengths, h0, c0, ctr);
    emissions_kernel<<<(SEQT * BATCH + 255) / 256, 256, 0, stream>>>(
        hs, hs + (size_t)SEQT * BATCH * HID, fc_W, fc_b, E);
    crf_kernel<<<BATCH, 32, 0, stream>>>(E, lengths, tags, trans, out);
}
